// LSTMNN_11897059409985
// MI455X (gfx1250) — matrix operand order repaired and checked by reading, not compiled, not run on hardware
//
#include <hip/hip_runtime.h>

// ---------------------------------------------------------------------------
// LSTM (2 layers, H=256, B=1024, T=128) + FC head on gfx1250.
// bf16 WMMA GEMMs with double-buffered GLOBAL_LOAD_ASYNC_TO_LDS staging.
// ---------------------------------------------------------------------------

#define Tlen 128
#define Bsz  1024
#define Hd   256
#define Gd   1024          // 4*H
#define Kd   512           // concat(x, h) depth
#define BM   128           // block tile M (batch rows)
#define BN   128           // block tile N (gate cols)
#define KC   32            // K chunk staged in LDS (one WMMA depth)
#define KCP  (KC + 8)      // padded LDS row (80 B: 16B-aligned, spreads banks)
#define NKB  (Kd / KC)     // 16 K chunks

typedef __attribute__((ext_vector_type(16))) __bf16 v16bf;
typedef __attribute__((ext_vector_type(8)))  float  v8f;

union FragBF { uint4 q[2]; v16bf v; };

__device__ __forceinline__ unsigned pack2bf(float a, float b) {
    __bf16 ba = (__bf16)a, bb = (__bf16)b;
    unsigned short ua = __builtin_bit_cast(unsigned short, ba);
    unsigned short ub = __builtin_bit_cast(unsigned short, bb);
    return (unsigned)ua | ((unsigned)ub << 16);
}

__device__ __forceinline__ float sigf(float x) { return 1.0f / (1.0f + __expf(-x)); }

// Async 16B global->LDS copy (GVS mode: SGPR base + per-lane byte voffset).
// Tracked by ASYNCcnt; completes in issue order.
__device__ __forceinline__ void async_cp16(unsigned lds_addr, unsigned voff,
                                           const void* sbase) {
    asm volatile("global_load_async_to_lds_b128 %0, %1, %2"
                 :: "v"(lds_addr), "v"(voff), "s"(sbase) : "memory");
}

// ---------------------------------------------------------------------------
// relu(x) f32 [B,T,256] -> bf16 time-major [T,B,256] (one-time, ~8us of HBM)
// ---------------------------------------------------------------------------
__global__ void prep_x(const float* __restrict__ x, __bf16* __restrict__ xbf) {
    const size_t o = ((size_t)blockIdx.x * 256 + threadIdx.x) * 8;
    const int t   = (int)(o >> 18);          // /(B*H) = /2^18
    const int rem = (int)(o & 262143);
    const int b   = rem >> 8;
    const int k   = rem & 255;
    const float* src = x + ((size_t)b * Tlen + t) * Hd + k;
    float4 f0 = ((const float4*)src)[0];
    float4 f1 = ((const float4*)src)[1];
    uint4 val;
    val.x = pack2bf(fmaxf(f0.x, 0.f), fmaxf(f0.y, 0.f));
    val.y = pack2bf(fmaxf(f0.z, 0.f), fmaxf(f0.w, 0.f));
    val.z = pack2bf(fmaxf(f1.x, 0.f), fmaxf(f1.y, 0.f));
    val.w = pack2bf(fmaxf(f1.z, 0.f), fmaxf(f1.w, 0.f));
    *(uint4*)(xbf + o) = val;
}

// ---------------------------------------------------------------------------
// Pack concat(w_ih, w_hh) -> bf16 [Gd][Kd] row-major; bias = b_ih + b_hh.
// ---------------------------------------------------------------------------
__global__ void pack_weights(const float* __restrict__ wih, const float* __restrict__ whh,
                             const float* __restrict__ bih, const float* __restrict__ bhh,
                             __bf16* __restrict__ Wcat, float* __restrict__ bias) {
    int idx = blockIdx.x * 256 + threadIdx.x;
    if (idx < Gd * Kd) {
        int n = idx >> 9;          // /512
        int k = idx & (Kd - 1);
        float v = (k < Hd) ? wih[n * Hd + k] : whh[n * Hd + (k - Hd)];
        Wcat[idx] = (__bf16)v;
    }
    if (idx < Gd) bias[idx] = bih[idx] + bhh[idx];
}

__global__ void init_state(float* __restrict__ c0, float* __restrict__ c1,
                           __bf16* __restrict__ hz) {
    int idx = blockIdx.x * 256 + threadIdx.x;   // B*H elements
    c0[idx] = 0.0f;
    c1[idx] = 0.0f;
    hz[idx] = (__bf16)0.0f;
}

// ---------------------------------------------------------------------------
// One timestep gate GEMM: gates[B,Gd] = A[B,Kd] x Wcat^T + bias
//   A[:, 0:256) = xsrc (bf16: relu'd x_t for layer0, hs0_t for layer1)
//   A[:,256:512) = h_prev (bf16)
// Double-buffered async global->LDS pipeline over 16 K-chunks of 32.
// ---------------------------------------------------------------------------
__global__ __launch_bounds__(256) void lstm_gemm_step(
        const __bf16* __restrict__ xsrc,   // [B,H] bf16
        const __bf16* __restrict__ hprev,  // [B,H] bf16
        const __bf16* __restrict__ Wcat,   // [Gd,Kd] bf16 row-major
        const float*  __restrict__ bias,   // [Gd]
        float*        __restrict__ gates)  // [B,Gd]
{
    __shared__ __bf16 sA[2][BM][KCP];
    __shared__ __bf16 sW[2][BN][KCP];

    const int tid  = threadIdx.x;
    const int bm   = blockIdx.y * BM;     // batch rows
    const int bn   = blockIdx.x * BN;     // gate cols
    const int wave = tid >> 5;
    const int lane = tid & 31;
    const int half = lane >> 4;
    const int l16  = lane & 15;
    const int wm   = wave >> 1;           // wave M tiles: 2*wm, 2*wm+1
    const int wn   = wave & 1;            // wave N tiles: 4*wn .. 4*wn+3

    v8f acc[2][4];
    const v8f vzero = {0.f, 0.f, 0.f, 0.f, 0.f, 0.f, 0.f, 0.f};
#pragma unroll
    for (int mi = 0; mi < 2; ++mi)
#pragma unroll
        for (int ni = 0; ni < 4; ++ni) acc[mi][ni] = vzero;

    // Issue async copies of one [128 x 32] A tile + W tile into stage st.
    // 512 16B-chunks per tile, 256 threads -> 2 A + 2 W copies per thread.
    auto issue = [&](int kb, int st) {
        const int kbase = kb * KC;                 // chunk never straddles x|h
        const __bf16* abase;
        unsigned aoff;
        if (kbase < Hd) { abase = xsrc;  aoff = (unsigned)kbase * 2u; }
        else            { abase = hprev; aoff = (unsigned)(kbase - Hd) * 2u; }
#pragma unroll
        for (int i = 0; i < 2; ++i) {
            const int cid = i * 256 + tid;          // 0..511
            const int row = cid >> 2;               // 4 chunks per row
            const int kcq = (cid & 3) * 8;
            const unsigned va = ((unsigned)(bm + row) * Hd + kcq) * 2u + aoff;
            const unsigned la = (unsigned)(uintptr_t)&sA[st][row][kcq];
            async_cp16(la, va, abase);
            const unsigned vw = ((unsigned)(bn + row) * Kd + (unsigned)kbase + kcq) * 2u;
            const unsigned lw = (unsigned)(uintptr_t)&sW[st][row][kcq];
            async_cp16(lw, vw, Wcat);
        }
    };

    issue(0, 0);

    for (int kb = 0; kb < NKB; ++kb) {
        const int st = kb & 1;
        if (kb + 1 < NKB) {
            issue(kb + 1, st ^ 1);   // prefetch next chunk into other buffer
            // async loads retire in order: <=4 outstanding => chunk kb landed
            asm volatile("s_wait_asynccnt 0x4" ::: "memory");
        } else {
            asm volatile("s_wait_asynccnt 0x0" ::: "memory");
        }
        __syncthreads();             // all waves' chunk-kb data visible

        FragBF af[2], bf[4];
#pragma unroll
        for (int mi = 0; mi < 2; ++mi) {
            const int row = (2 * wm + mi) * 16 + l16;
            af[mi].q[0] = *(const uint4*)&sA[st][row][half * 8];       // K 0..7 / 8..15
            af[mi].q[1] = *(const uint4*)&sA[st][row][16 + half * 8];  // K 16..23 / 24..31
        }
#pragma unroll
        for (int ni = 0; ni < 4; ++ni) {
            const int row = (4 * wn + ni) * 16 + l16;
            bf[ni].q[0] = *(const uint4*)&sW[st][row][half * 8];       // K 0..7 / 8..15
            bf[ni].q[1] = *(const uint4*)&sW[st][row][16 + half * 8];  // K 16..23 / 24..31
        }
#pragma unroll
        for (int mi = 0; mi < 2; ++mi)
#pragma unroll
            for (int ni = 0; ni < 4; ++ni)
                acc[mi][ni] = __builtin_amdgcn_wmma_f32_16x16x32_bf16(
                    false, af[mi].v, false, bf[ni].v,
                    (short)0, acc[mi][ni], false, false);

        __syncthreads();             // buffer st free before it's refilled (kb+2)
    }

    // ---- epilogue: C frag VGPR r -> M = r + half*8, N = l16 (ISA 7.12.2)
#pragma unroll
    for (int mi = 0; mi < 2; ++mi)
#pragma unroll
        for (int ni = 0; ni < 4; ++ni) {
            const int gm = bm + (2 * wm + mi) * 16 + half * 8;
            const int gn = bn + (4 * wn + ni) * 16 + l16;
            const float bv = bias[gn];
#pragma unroll
            for (int r = 0; r < 8; ++r)
                gates[(size_t)(gm + r) * Gd + gn] = acc[mi][ni][r] + bv;
        }
}

// ---------------------------------------------------------------------------
// Elementwise LSTM cell update. gates layout per row: [i | f | g | o] (Gd).
// ---------------------------------------------------------------------------
__global__ void lstm_cell(const float* __restrict__ gates, float* __restrict__ c,
                          __bf16* __restrict__ hb, float* __restrict__ hf) {
    const int idx = blockIdx.x * 256 + threadIdx.x;  // 0..B*H
    const int b = idx >> 8;
    const int j = idx & 255;
    const float* g = gates + (size_t)b * Gd;
    const float gi = g[j], gf = g[256 + j], gg = g[512 + j], go = g[768 + j];
    const float cn = sigf(gf) * c[idx] + sigf(gi) * tanhf(gg);
    const float h  = sigf(go) * tanhf(cn);
    c[idx]  = cn;
    hb[idx] = (__bf16)h;
    hf[idx] = h;
}

// ---------------------------------------------------------------------------
// FC head: out[b] = sigmoid(fc2(relu(fc1(h_last[b]))))
// ---------------------------------------------------------------------------
__global__ __launch_bounds__(128) void fc_head(const float* __restrict__ hlast,
        const float* __restrict__ w1, const float* __restrict__ b1,
        const float* __restrict__ w2, const float* __restrict__ b2,
        float* __restrict__ out) {
    __shared__ float sm[128];
    const int b = blockIdx.x;
    const int j = threadIdx.x;
    const float* hv = hlast + (size_t)b * Hd;
    const float* wr = w1 + (size_t)j * Hd;
    float acc = b1[j];
#pragma unroll 4
    for (int k = 0; k < Hd; k += 4) {
        float4 h4 = *(const float4*)(hv + k);
        float4 w4 = *(const float4*)(wr + k);
        acc += h4.x * w4.x + h4.y * w4.y + h4.z * w4.z + h4.w * w4.w;
    }
    acc = fmaxf(acc, 0.0f);
    sm[j] = acc * w2[j];
    __syncthreads();
    for (int s = 64; s > 0; s >>= 1) {
        if (j < s) sm[j] += sm[j + s];
        __syncthreads();
    }
    if (j == 0) out[b] = sigf(sm[0] + b2[0]);
}

// ---------------------------------------------------------------------------
// Host-side launch sequence (all on `stream`, graph-capture safe).
// ---------------------------------------------------------------------------
extern "C" void kernel_launch(void* const* d_in, const int* in_sizes, int n_in,
                              void* d_out, int out_size, void* d_ws, size_t ws_size,
                              hipStream_t stream) {
    (void)in_sizes; (void)n_in; (void)out_size; (void)ws_size;

    const float* x     = (const float*)d_in[0];
    const float* wih0  = (const float*)d_in[1];
    const float* whh0  = (const float*)d_in[2];
    const float* bih0  = (const float*)d_in[3];
    const float* bhh0  = (const float*)d_in[4];
    const float* wih1  = (const float*)d_in[5];
    const float* whh1  = (const float*)d_in[6];
    const float* bih1  = (const float*)d_in[7];
    const float* bhh1  = (const float*)d_in[8];
    const float* fc1w  = (const float*)d_in[9];
    const float* fc1b  = (const float*)d_in[10];
    const float* fc2w  = (const float*)d_in[11];
    const float* fc2b  = (const float*)d_in[12];

    // Workspace layout (all offsets 256B-aligned), ~140 MiB total.
    char* ws = (char*)d_ws;
    size_t off = 0;
    __bf16* xbf   = (__bf16*)(ws + off); off += (size_t)Tlen * Bsz * Hd * 2;  // 64 MiB
    __bf16* hs0   = (__bf16*)(ws + off); off += (size_t)Tlen * Bsz * Hd * 2;  // 64 MiB
    __bf16* Wcat0 = (__bf16*)(ws + off); off += (size_t)Gd * Kd * 2;          // 1 MiB
    __bf16* Wcat1 = (__bf16*)(ws + off); off += (size_t)Gd * Kd * 2;          // 1 MiB
    float*  bias0 = (float*) (ws + off); off += (size_t)Gd * 4;
    float*  bias1 = (float*) (ws + off); off += (size_t)Gd * 4;
    float*  gates = (float*) (ws + off); off += (size_t)Bsz * Gd * 4;         // 4 MiB
    float*  c0    = (float*) (ws + off); off += (size_t)Bsz * Hd * 4;
    float*  c1    = (float*) (ws + off); off += (size_t)Bsz * Hd * 4;
    __bf16* h1a   = (__bf16*)(ws + off); off += (size_t)Bsz * Hd * 2;
    __bf16* h1b   = (__bf16*)(ws + off); off += (size_t)Bsz * Hd * 2;
    __bf16* hzero = (__bf16*)(ws + off); off += (size_t)Bsz * Hd * 2;
    float*  hlast = (float*) (ws + off); off += (size_t)Bsz * Hd * 4;

    // --- prep: relu+cvt x, pack weights/biases (bf16), zero state
    prep_x<<<(Tlen * Bsz * Hd) / (8 * 256), 256, 0, stream>>>(x, xbf);
    pack_weights<<<(Gd * Kd + 255) / 256, 256, 0, stream>>>(wih0, whh0, bih0, bhh0, Wcat0, bias0);
    pack_weights<<<(Gd * Kd + 255) / 256, 256, 0, stream>>>(wih1, whh1, bih1, bhh1, Wcat1, bias1);
    init_state<<<(Bsz * Hd) / 256, 256, 0, stream>>>(c0, c1, hzero);

    const dim3 ggrid(Gd / BN, Bsz / BM);  // 8 x 8

    // --- layer 0: x path from xbf; h_prev comes from hs0[t-1]
    for (int t = 0; t < Tlen; ++t) {
        const __bf16* hprev = (t == 0) ? hzero : hs0 + (size_t)(t - 1) * Bsz * Hd;
        lstm_gemm_step<<<ggrid, 256, 0, stream>>>(
            xbf + (size_t)t * Bsz * Hd, hprev, Wcat0, bias0, gates);
        lstm_cell<<<(Bsz * Hd) / 256, 256, 0, stream>>>(
            gates, c0, hs0 + (size_t)t * Bsz * Hd, hlast);
    }

    // --- layer 1: input path is bf16 hs0[t]; ping-pong h state
    for (int t = 0; t < Tlen; ++t) {
        const __bf16* hprev = (t == 0) ? hzero : ((t & 1) ? h1a : h1b);
        __bf16* hout = (t & 1) ? h1b : h1a;
        lstm_gemm_step<<<ggrid, 256, 0, stream>>>(
            hs0 + (size_t)t * Bsz * Hd, hprev, Wcat1, bias1, gates);
        lstm_cell<<<(Bsz * Hd) / 256, 256, 0, stream>>>(gates, c1, hout, hlast);
    }

    // --- FC head on final h (f32)
    fc_head<<<Bsz, 128, 0, stream>>>(hlast, fc1w, fc1b, fc2w, fc2b, (float*)d_out);
}
